// RAG4DyG_9672266351243
// MI455X (gfx1250) — compile-verified, matrix operand order checked by reading, not run on hardware
//
#include <hip/hip_runtime.h>
#include <hip/hip_bf16.h>
#include <math.h>

typedef __attribute__((ext_vector_type(16))) __bf16 v16bf;
typedef __attribute__((ext_vector_type(8)))  __bf16 v8bf;
typedef __attribute__((ext_vector_type(8)))  float  v8f;

#define B_DIM   4096
#define N_DIM   100
#define NPAD    112        // 7 tiles of 16 rows
#define NODE_F  128
#define D_DIM   64
#define T_DIM   8

// packed bf16 weight fragments in d_ws (each fragment: 32 lanes x 16 bf16 = 512 elems)
#define OFF_FEAT 0            // W_feat: 4 ktiles x 4 ntiles = 16 frags
#define OFF_GCN  (16 * 512)   // W_gcn : 2 x 4 = 8 frags
#define OFF_OUT  (24 * 512)   // W_out : 2 x 4 = 8 frags
#define PACK_TOT (32 * 512)

__device__ __forceinline__ v8f wmma_bf16_f32(v16bf a, v16bf b, v8f c) {
  // D = A(16x32 bf16) * B(32x16 bf16) + C(16x16 f32)
  return __builtin_amdgcn_wmma_f32_16x16x32_bf16(false, a, false, b, (short)0, c,
                                                 false, false);
}

__device__ __forceinline__ v16bf cat8(v8bf lo, v8bf hi) {
  return __builtin_shufflevector(lo, hi, 0, 1, 2, 3, 4, 5, 6, 7,
                                         8, 9, 10, 11, 12, 13, 14, 15);
}

// ---------------------------------------------------------------------------
// Kernel 0: convert W_feat/W_gcn/W_out (fp32, [K][64] row-major) to bf16 in
// WMMA B-fragment order: frag f=(kt,nt); lane L holds col N=nt*16+(L&15),
// element e holds K = kt*32 + ((e<8)?e:e+8) + ((L>=16)?8:0).
// ---------------------------------------------------------------------------
__global__ __launch_bounds__(256)
void pack_weights(const float* __restrict__ Wf, const float* __restrict__ Wg,
                  const float* __restrict__ Wo, __bf16* __restrict__ ws) {
  int idx = blockIdx.x * 256 + threadIdx.x;
  if (idx >= PACK_TOT) return;
  int f    = idx >> 9;       // global fragment 0..31
  int r    = idx & 511;
  int lane = r >> 4;
  int e    = r & 15;
  int kk = ((e < 8) ? e : e + 8) + ((lane >= 16) ? 8 : 0);
  int nn = lane & 15;
  const float* W;
  int fl;
  if (f < 16)      { W = Wf; fl = f; }
  else if (f < 24) { W = Wg; fl = f - 16; }
  else             { W = Wo; fl = f - 24; }
  int kt = fl >> 2, nt = fl & 3;
  ws[idx] = (__bf16)W[(kt * 32 + kk) * 64 + nt * 16 + nn];
}

// ---------------------------------------------------------------------------
// Kernel 1: full per-sample fused pipeline. One block (8 waves) per sample b.
// ---------------------------------------------------------------------------
__global__ __launch_bounds__(256)
void rag_main(const int* __restrict__ nbr_ids, const int* __restrict__ edge_ids,
              const float* __restrict__ times, const int* __restrict__ dst_ids,
              const float* __restrict__ node_raw, const float* __restrict__ edge_raw,
              const float* __restrict__ b_feat, const float* __restrict__ W_edge,
              const float* __restrict__ b_edge, const float* __restrict__ W_time,
              const float* __restrict__ b_time, const float* __restrict__ W_struct,
              const float* __restrict__ b_struct, const float* __restrict__ w_tenc,
              const float* __restrict__ b_tenc, const float* __restrict__ ln_g,
              const float* __restrict__ ln_b, const float* __restrict__ b_gcn,
              const float* __restrict__ W_out, const float* __restrict__ b_out,
              const __bf16* __restrict__ wpack, float* __restrict__ out) {
  // Region X is time-aliased: gathered bf16 rows -> fused f32 -> h f32.
  __shared__ __align__(16) unsigned char smemX[NPAD * NODE_F * 2];  // 28672 B
  __bf16* nbrS  = (__bf16*)smemX;   // [NPAD][128] bf16
  float*  fusedS = (float*)smemX;   // [NPAD][64]  f32 (after WMMA1)
  float*  hS     = (float*)smemX;   // [NPAD][64]  f32 (after WMMA2)
  __shared__ __align__(16) __bf16 fusedBF[NPAD * D_DIM];            // 14336 B
  __shared__ float tencS[NPAD * T_DIM];
  __shared__ float edgeV[NPAD];
  __shared__ float extraV[NPAD];    // co + sim (both multiply W_struct)
  __shared__ int   idsS[NPAD];
  __shared__ float poolS[4 * D_DIM];
  __shared__ int   vS;

  const int tid    = threadIdx.x;
  const int b      = blockIdx.x;
  const int lane   = tid & 31;
  const int lane15 = lane & 15;
  const int laneHi = lane >> 4;     // 0 or 1
  const int wave   = tid >> 5;      // 0..7

  if (tid == 0) vS = 0;
  __syncthreads();

  // ---- Stage 0: per-row scalar staging -----------------------------------
  if (tid < NPAD) {
    int n = tid;
    int id = -1;
    float tm = 0.f, ec = 0.f, ex = 0.f;
    if (n < N_DIM) {
      id = nbr_ids[b * N_DIM + n];
      tm = times[b * N_DIM + n];
      int eid = edge_ids[b * N_DIM + n];
      ec = edge_raw[eid * NODE_F];                    // column 0
      int dstid = dst_ids[b];
      int skill_dst = (int)node_raw[dstid * NODE_F];
      int skill_n   = (int)node_raw[id * NODE_F];
      ex = (id == dstid ? 1.f : 0.f) + (skill_n == skill_dst ? 1.f : 0.f);
      if (id > 0) atomicAdd(&vS, 1);
    }
    idsS[n] = id;
    edgeV[n] = ec;
    extraV[n] = ex;
#pragma unroll
    for (int t = 0; t < T_DIM; ++t)
      tencS[n * T_DIM + t] = (n < N_DIM) ? cosf(tm * w_tenc[t] + b_tenc[t]) : 0.f;
  }
  __syncthreads();

  // ---- Stage 1a: gather neighbor rows -> bf16 tile in LDS ----------------
  for (int idx = tid; idx < NPAD * NODE_F; idx += 256) {
    int n = idx >> 7, c = idx & 127;
    int id = idsS[n];
    float v = (id >= 0) ? node_raw[id * NODE_F + c] : 0.f;
    nbrS[idx] = (__bf16)v;
  }
  __syncthreads();

  // ---- Stage 1b: WMMA  nbr[112x128] @ W_feat[128x64] ---------------------
  v8f accv[4];
#pragma unroll
  for (int s = 0; s < 4; ++s) {
    int f = wave + s * 8;              // uniform per wave
    if (f < 28) {
      int rt = f >> 2, ct = f & 3;
      v8f acc = {};
#pragma unroll
      for (int kt = 0; kt < 4; ++kt) {
        const __bf16* ap = nbrS + (rt * 16 + lane15) * NODE_F + kt * 32 + laneHi * 8;
        v16bf a  = cat8(*(const v8bf*)ap, *(const v8bf*)(ap + 16));
        v16bf bb = *(const v16bf*)(wpack + OFF_FEAT + (kt * 4 + ct) * 512 + lane * 16);
        acc = wmma_bf16_f32(a, bb, acc);
      }
      accv[s] = acc;
    }
  }
  __syncthreads();   // all reads of nbrS done; region X may be overwritten

  // ---- Stage 1c: fold rank-1 extras + biases, store fused f32 ------------
#pragma unroll
  for (int s = 0; s < 4; ++s) {
    int f = wave + s * 8;
    if (f < 28) {
      int rt = f >> 2, ct = f & 3;
      int d = ct * 16 + lane15;        // column fixed per lane for C layout
      float wedge = W_edge[d];
      float wst   = W_struct[d];
      float sb    = b_feat[d] + b_edge[d] + 2.f * b_struct[d] + b_time[d];
      float wt[T_DIM];
#pragma unroll
      for (int t = 0; t < T_DIM; ++t) wt[t] = W_time[t * D_DIM + d];
      v8f acc = accv[s];
#pragma unroll
      for (int r = 0; r < 8; ++r) {
        int n = rt * 16 + r + laneHi * 8;
        float val = acc[r] + sb + edgeV[n] * wedge + extraV[n] * wst;
#pragma unroll
        for (int t = 0; t < T_DIM; ++t) val += tencS[n * T_DIM + t] * wt[t];
        fusedS[n * D_DIM + d] = val;
      }
    }
  }
  __syncthreads();

  // ---- Stage 2: layernorm (f32) -> bf16 ----------------------------------
  for (int n = tid; n < NPAD; n += 256) {
    float m = 0.f;
    for (int d2 = 0; d2 < D_DIM; ++d2) m += fusedS[n * D_DIM + d2];
    m *= (1.f / 64.f);
    float var = 0.f;
    for (int d2 = 0; d2 < D_DIM; ++d2) {
      float x = fusedS[n * D_DIM + d2] - m;
      var += x * x;
    }
    var *= (1.f / 64.f);
    float rs = rsqrtf(var + 1e-5f);
    for (int d2 = 0; d2 < D_DIM; ++d2) {
      float y = (n < N_DIM) ? (fusedS[n * D_DIM + d2] - m) * rs * ln_g[d2] + ln_b[d2]
                            : 0.f;
      fusedBF[n * D_DIM + d2] = (__bf16)y;
    }
  }
  __syncthreads();   // fusedS dead; region X free for h

  // ---- Stage 3: WMMA  fused[112x64] @ W_gcn[64x64] -----------------------
#pragma unroll
  for (int s = 0; s < 4; ++s) {
    int f = wave + s * 8;
    if (f < 28) {
      int rt = f >> 2, ct = f & 3;
      v8f acc = {};
#pragma unroll
      for (int kt = 0; kt < 2; ++kt) {
        const __bf16* ap = fusedBF + (rt * 16 + lane15) * D_DIM + kt * 32 + laneHi * 8;
        v16bf a  = cat8(*(const v8bf*)ap, *(const v8bf*)(ap + 16));
        v16bf bb = *(const v16bf*)(wpack + OFF_GCN + (kt * 4 + ct) * 512 + lane * 16);
        acc = wmma_bf16_f32(a, bb, acc);
      }
      accv[s] = acc;
    }
  }
  __syncthreads();
#pragma unroll
  for (int s = 0; s < 4; ++s) {
    int f = wave + s * 8;
    if (f < 28) {
      int rt = f >> 2, ct = f & 3;
      int d = ct * 16 + lane15;
      v8f acc = accv[s];
#pragma unroll
      for (int r = 0; r < 8; ++r) {
        int n = rt * 16 + r + laneHi * 8;
        hS[n * D_DIM + d] = acc[r];
      }
    }
  }
  __syncthreads();

  // ---- Stage 4: chain-GCN combine + ReLU + mean pool ---------------------
  {
    int v = vS;
    int g = tid >> 6, d = tid & 63;
    float bg = b_gcn[d];
    float sum = 0.f;
    for (int n = g; n < N_DIM; n += 4) {
      int   hin  = (n >= 1 && n < v) ? 1 : 0;
      float deg  = 1.f + (float)hin;
      float dinv = hin ? 0.70710678118654752f : 1.f;
      int   hinp = ((n - 1) >= 1 && (n - 1) < v) ? 1 : 0;
      float dinvp = hinp ? 0.70710678118654752f : 1.f;
      float hp = (n >= 1) ? hS[(n - 1) * D_DIM + d] : 0.f;
      float o  = hS[n * D_DIM + d] / deg + (float)hin * dinvp * dinv * hp + bg;
      sum += fmaxf(o, 0.f);
    }
    poolS[g * 64 + d] = sum;
  }
  __syncthreads();
  if (tid < 64) {
    float p = (poolS[tid] + poolS[64 + tid] + poolS[128 + tid] + poolS[192 + tid])
              * (1.f / (float)N_DIM);
    poolS[tid] = p;
  }
  __syncthreads();

  // ---- Stage 5: src_emb = pooled @ W_out + b_out -------------------------
  if (tid < 64) {
    float acc = b_out[tid];
    for (int k = 0; k < 64; ++k) acc += poolS[k] * W_out[k * 64 + tid];
    out[b * 64 + tid] = acc;
  }
}

// ---------------------------------------------------------------------------
// Kernel 2: dst_emb = (node_raw[dst] @ W_feat + b_feat) @ W_out + b_out
// One wave per 16 dst rows.
// ---------------------------------------------------------------------------
__global__ __launch_bounds__(32)
void rag_dst(const int* __restrict__ dst_ids, const float* __restrict__ node_raw,
             const float* __restrict__ b_feat, const float* __restrict__ b_out,
             const __bf16* __restrict__ wpack, float* __restrict__ out) {
  __shared__ __align__(16) __bf16 rowsBF[16 * NODE_F];
  __shared__ __align__(16) __bf16 tmpBF[16 * D_DIM];
  const int tid = threadIdx.x;
  const int blk = blockIdx.x;
  const int lane15 = tid & 15, laneHi = tid >> 4;

  for (int idx = tid; idx < 16 * NODE_F; idx += 32) {
    int r = idx >> 7, c = idx & 127;
    int id = dst_ids[blk * 16 + r];
    rowsBF[idx] = (__bf16)node_raw[id * NODE_F + c];
  }
  __syncthreads();

#pragma unroll
  for (int ct = 0; ct < 4; ++ct) {
    v8f acc = {};
#pragma unroll
    for (int kt = 0; kt < 4; ++kt) {
      const __bf16* ap = rowsBF + lane15 * NODE_F + kt * 32 + laneHi * 8;
      v16bf a  = cat8(*(const v8bf*)ap, *(const v8bf*)(ap + 16));
      v16bf bb = *(const v16bf*)(wpack + OFF_FEAT + (kt * 4 + ct) * 512 + tid * 16);
      acc = wmma_bf16_f32(a, bb, acc);
    }
    int d = ct * 16 + lane15;
    float bf = b_feat[d];
#pragma unroll
    for (int r = 0; r < 8; ++r) {
      int m = r + laneHi * 8;
      tmpBF[m * D_DIM + d] = (__bf16)(acc[r] + bf);
    }
  }
  __syncthreads();

#pragma unroll
  for (int ct = 0; ct < 4; ++ct) {
    v8f acc = {};
#pragma unroll
    for (int kt = 0; kt < 2; ++kt) {
      const __bf16* ap = tmpBF + lane15 * D_DIM + kt * 32 + laneHi * 8;
      v16bf a  = cat8(*(const v8bf*)ap, *(const v8bf*)(ap + 16));
      v16bf bb = *(const v16bf*)(wpack + OFF_OUT + (kt * 4 + ct) * 512 + tid * 16);
      acc = wmma_bf16_f32(a, bb, acc);
    }
    int d = ct * 16 + lane15;
    float bo = b_out[d];
#pragma unroll
    for (int r = 0; r < 8; ++r) {
      int grow = blk * 16 + r + laneHi * 8;
      out[B_DIM * 64 + grow * 64 + d] = acc[r] + bo;
    }
  }
}

// ---------------------------------------------------------------------------
extern "C" void kernel_launch(void* const* d_in, const int* in_sizes, int n_in,
                              void* d_out, int out_size, void* d_ws, size_t ws_size,
                              hipStream_t stream) {
  const int*   nbr_ids  = (const int*)d_in[0];
  const int*   edge_ids = (const int*)d_in[1];
  const float* times    = (const float*)d_in[2];
  const int*   dst_ids  = (const int*)d_in[3];
  const float* node_raw = (const float*)d_in[4];
  const float* edge_raw = (const float*)d_in[5];
  const float* W_feat   = (const float*)d_in[6];
  const float* b_feat   = (const float*)d_in[7];
  const float* W_edge   = (const float*)d_in[8];
  const float* b_edge   = (const float*)d_in[9];
  const float* W_time   = (const float*)d_in[10];
  const float* b_time   = (const float*)d_in[11];
  const float* W_struct = (const float*)d_in[12];
  const float* b_struct = (const float*)d_in[13];
  const float* w_tenc   = (const float*)d_in[14];
  const float* b_tenc   = (const float*)d_in[15];
  const float* ln_g     = (const float*)d_in[16];
  const float* ln_b     = (const float*)d_in[17];
  const float* W_gcn    = (const float*)d_in[18];
  const float* b_gcn    = (const float*)d_in[19];
  const float* W_out    = (const float*)d_in[20];
  const float* b_out    = (const float*)d_in[21];
  float* out = (float*)d_out;
  __bf16* wpack = (__bf16*)d_ws;

  pack_weights<<<PACK_TOT / 256, 256, 0, stream>>>(W_feat, W_gcn, W_out, wpack);
  rag_main<<<B_DIM, 256, 0, stream>>>(nbr_ids, edge_ids, times, dst_ids, node_raw,
                                      edge_raw, b_feat, W_edge, b_edge, W_time,
                                      b_time, W_struct, b_struct, w_tenc, b_tenc,
                                      ln_g, ln_b, b_gcn, W_out, b_out, wpack, out);
  rag_dst<<<B_DIM / 16, 32, 0, stream>>>(dst_ids, node_raw, b_feat, b_out, wpack, out);
}
